// MultiModalGNN_80487687127290
// MI455X (gfx1250) — matrix operand order repaired and checked by reading, not compiled, not run on hardware
//
#include <hip/hip_runtime.h>

// ---------------------------------------------------------------------------
// MI455X (gfx1250) implementation of a 2-layer GCN + BN/ReLU + 2 MLP heads.
//
// Roofline reasoning:
//  * Dense GEMMs (~10 GFLOP) -> v_wmma_f32_16x16x32_bf16 (bf16 in, f32 acc).
//    K=128 => 4 WMMAs per 16x16 tile; weights pre-swizzled to a contiguous-k
//    B-matrix lane layout in LDS, read back as ds_load_b128 pairs.
//  * bf16 packing uses __builtin_convertvector so the backend can emit native
//    packed f32->bf16 converts instead of a 4-op manual RNE sequence.
//  * Edge aggregation (2 x 1.6M edges x 1KB) dominates bytes, but the 51MB
//    gathered matrix fits in the 192MB L2, so gathers + f32 atomic scatters
//    stay on-chip. One wave32 per edge: lane l moves float4 l (coalesced
//    512B per edge), 4x global_atomic_add_f32 per lane.
//  * Store tails: only the last block is partial, so stores take a
//    wave-uniform unguarded fast path except at the boundary.
// ---------------------------------------------------------------------------

typedef __attribute__((ext_vector_type(16))) __bf16 v16bf;
typedef __attribute__((ext_vector_type(2)))  __bf16 bf16x2;
typedef __attribute__((ext_vector_type(8)))  float  v8f;
typedef __attribute__((ext_vector_type(2)))  float  f32x2;

union FragBF {
  unsigned int u[8];
  uint4        q[2];
  v16bf        v;
};

__device__ __forceinline__ unsigned int pack2bf(float lo, float hi) {
  f32x2 f = {lo, hi};
  bf16x2 b = __builtin_convertvector(f, bf16x2);   // native v_cvt path on gfx1250
  union { bf16x2 b; unsigned int u; } c;
  c.b = b;
  return c.u;
}
__device__ __forceinline__ unsigned short f2bf(float f) {
  union { __bf16 b; unsigned short u; } c;
  c.b = (__bf16)f;
  return c.u;
}
__device__ __forceinline__ float bf2f(unsigned short b) {
  return __uint_as_float(((unsigned int)b) << 16);
}

// ---------------------------------------------------------------------------
// Graph normalization kernels
// ---------------------------------------------------------------------------
__global__ __launch_bounds__(256) void gnn_deg_init(float* __restrict__ deg, int N) {
  int i = blockIdx.x * 256 + threadIdx.x;
  if (i < N) deg[i] = 1.0f;                       // self-loop weight
}

__global__ __launch_bounds__(256) void gnn_deg_acc(const long long* __restrict__ ei,
                                                   const float* __restrict__ ew,
                                                   float* __restrict__ deg, int E) {
  int e = blockIdx.x * 256 + threadIdx.x;
  if (e < E) atomicAdd(&deg[(long long)ei[(long long)E + e]], ew[e]);
}

__global__ __launch_bounds__(256) void gnn_dinv(float* __restrict__ deg, int N) {
  int i = blockIdx.x * 256 + threadIdx.x;
  if (i < N) {
    float d = deg[i];
    deg[i] = (d > 0.f) ? rsqrtf(d) : 0.f;
  }
}

__global__ __launch_bounds__(256) void gnn_norm(const long long* __restrict__ ei,
                                                const float* __restrict__ ew,
                                                const float* __restrict__ dinv,
                                                float* __restrict__ nrm, int E) {
  int e = blockIdx.x * 256 + threadIdx.x;
  if (e < E) {
    long long r = ei[e];
    long long c = ei[(long long)E + e];
    nrm[e] = dinv[r] * ew[e] * dinv[c];
  }
}

// ---------------------------------------------------------------------------
// Dense 128-wide GEMM: C[N,128] = A[N,128] @ W[128,128] via bf16 WMMA.
// Block = 256 threads = 8 waves; each block does a 128-row strip, each wave a
// 16-row strip across all eight 16-column tiles. W is swizzled once into LDS
// in this code's contiguous-k B-matrix layout (lane n = L&15, k = 32*ks + 16*half + 2j).
// ---------------------------------------------------------------------------
__global__ __launch_bounds__(256) void gnn_gemm128(const float* __restrict__ A,
                                                   const float* __restrict__ W,
                                                   float* __restrict__ C, int Nrows) {
  __shared__ __align__(16) unsigned int sB[8192];   // 8 ntiles x 4 ksteps x 32 lanes x 8 dw
  const int tid = threadIdx.x;
  for (int d = tid; d < 8192; d += 256) {
    const int j  = d & 7;
    const int ln = (d >> 3) & 31;
    const int r  = d >> 8;
    const int ks = r & 3;
    const int nt = r >> 2;
    const int n  = nt * 16 + (ln & 15);
    const int k  = ks * 32 + ((ln >> 4) << 4) + 2 * j;
    sB[d] = pack2bf(W[k * 128 + n], W[(k + 1) * 128 + n]);
  }
  __syncthreads();

  const int wave = tid >> 5;
  const int lane = tid & 31;
  const int m    = lane & 15;
  const int half = lane >> 4;
  const int row0 = blockIdx.x * 128 + wave * 16;
  int row = row0 + m;
  if (row > Nrows - 1) row = Nrows - 1;             // clamp: keep EXEC full for WMMA
  const float2* arow = reinterpret_cast<const float2*>(A + (long long)row * 128);

  // A fragments for all 4 K-steps (A layout: k = 32*ks + 16*half + 2*j)
  FragBF a[4];
#pragma unroll
  for (int ks = 0; ks < 4; ++ks) {
#pragma unroll
    for (int j = 0; j < 8; ++j) {
      const int k = ks * 32 + half * 16 + 2 * j;
      float2 p = arow[k >> 1];
      a[ks].u[j] = pack2bf(p.x, p.y);
    }
  }

  const bool full = (row0 + 16) <= Nrows;           // wave-uniform fast path
#pragma unroll
  for (int nt = 0; nt < 8; ++nt) {
    v8f acc = {0.f, 0.f, 0.f, 0.f, 0.f, 0.f, 0.f, 0.f};
#pragma unroll
    for (int ks = 0; ks < 4; ++ks) {
      FragBF b;
      const uint4* p = reinterpret_cast<const uint4*>(&sB[((nt * 4 + ks) * 32 + lane) * 8]);
      b.q[0] = p[0];
      b.q[1] = p[1];
      acc = __builtin_amdgcn_wmma_f32_16x16x32_bf16(false, a[ks].v, false, b.v,
                                                    (short)0, acc, false, false);
    }
    const int n = nt * 16 + m;
    float* cb = C + (long long)(row0 + half * 8) * 128 + n;
    if (full) {
#pragma unroll
      for (int vi = 0; vi < 8; ++vi) cb[vi * 128] = acc[vi];
    } else {
#pragma unroll
      for (int vi = 0; vi < 8; ++vi) {
        if (row0 + vi + half * 8 < Nrows) cb[vi * 128] = acc[vi];
      }
    }
  }
}

// ---------------------------------------------------------------------------
// acc[i,:] = dinv[i]^2 * xw[i,:]   (self-loop term initializes accumulator)
// ---------------------------------------------------------------------------
__global__ __launch_bounds__(256) void gnn_selfinit(const float* __restrict__ xw,
                                                    const float* __restrict__ dinv,
                                                    float* __restrict__ acc, int nvec) {
  int i = blockIdx.x * 256 + threadIdx.x;           // over N*32 float4s
  if (i >= nvec) return;
  const int node = i >> 5;
  const float d = dinv[node];
  const float s = d * d;
  float4 v = reinterpret_cast<const float4*>(xw)[i];
  v.x *= s; v.y *= s; v.z *= s; v.w *= s;
  reinterpret_cast<float4*>(acc)[i] = v;
}

// ---------------------------------------------------------------------------
// Edge scatter-add: one wave per edge, lane l moves float4 element l.
// Gather source (51MB) is L2-resident; atomics land in L2.
// ---------------------------------------------------------------------------
__global__ __launch_bounds__(256) void gnn_edge_agg(const float* __restrict__ src,
                                                    const float* __restrict__ nrm,
                                                    const long long* __restrict__ ei,
                                                    float* __restrict__ acc, int E) {
  const int e = blockIdx.x * 8 + (threadIdx.x >> 5);
  if (e >= E) return;                               // wave-uniform
  const int lane = threadIdx.x & 31;
  const long long row = ei[e];
  const long long col = ei[(long long)E + e];
  const float w = nrm[e];
  float4 v = reinterpret_cast<const float4*>(src)[row * 32 + lane];
  float* dst = acc + col * 128 + lane * 4;
  atomicAdd(dst + 0, w * v.x);
  atomicAdd(dst + 1, w * v.y);
  atomicAdd(dst + 2, w * v.z);
  atomicAdd(dst + 3, w * v.w);
}

// ---------------------------------------------------------------------------
// dst = relu( (src + bias - mean) * rsqrt(var+eps) * gamma + beta ), float4.
// ---------------------------------------------------------------------------
__global__ __launch_bounds__(256) void gnn_bnrelu(const float* __restrict__ src,
                                                  const float* __restrict__ bias,
                                                  const float* __restrict__ gamma,
                                                  const float* __restrict__ beta,
                                                  const float* __restrict__ mean,
                                                  const float* __restrict__ var,
                                                  float* __restrict__ dst, int nvec) {
  int i = blockIdx.x * 256 + threadIdx.x;
  if (i >= nvec) return;
  const int j4 = i & 31;
  float4 v  = reinterpret_cast<const float4*>(src)[i];
  float4 bb = reinterpret_cast<const float4*>(bias)[j4];
  float4 gg = reinterpret_cast<const float4*>(gamma)[j4];
  float4 be = reinterpret_cast<const float4*>(beta)[j4];
  float4 mn = reinterpret_cast<const float4*>(mean)[j4];
  float4 vr = reinterpret_cast<const float4*>(var)[j4];
  float4 o;
  o.x = fmaxf(0.f, (v.x + bb.x - mn.x) * rsqrtf(vr.x + 1e-5f) * gg.x + be.x);
  o.y = fmaxf(0.f, (v.y + bb.y - mn.y) * rsqrtf(vr.y + 1e-5f) * gg.y + be.y);
  o.z = fmaxf(0.f, (v.z + bb.z - mn.z) * rsqrtf(vr.z + 1e-5f) * gg.z + be.z);
  o.w = fmaxf(0.f, (v.w + bb.w - mn.w) * rsqrtf(vr.w + 1e-5f) * gg.w + be.w);
  reinterpret_cast<float4*>(dst)[i] = o;
}

// ---------------------------------------------------------------------------
// Fused heads: cls = relu(H@Wc1+bc1)@Wc2+bc2, key = relu(H@Wk1+bk1)@Wk2+bk2.
// Stage-1 via WMMA; relu'd 16x64 tile staged in per-wave LDS (bf16, A-layout
// readable), stage-2 class via WMMA, scalar key head via a 64-term VALU dot.
// ---------------------------------------------------------------------------
__global__ __launch_bounds__(256) void gnn_heads(const float* __restrict__ H,
    const float* __restrict__ Wc1, const float* __restrict__ bc1,
    const float* __restrict__ Wc2, const float* __restrict__ bc2,
    const float* __restrict__ Wk1, const float* __restrict__ bk1,
    const float* __restrict__ Wk2, const float* __restrict__ bk2,
    float* __restrict__ cls, float* __restrict__ key, int Nrows) {
  __shared__ __align__(16) unsigned int sC1[4096];  // Wc1 128x64 swizzled
  __shared__ __align__(16) unsigned int sK1[4096];  // Wk1 128x64 swizzled
  __shared__ __align__(16) unsigned int sC2[1024];  // Wc2 64x32 swizzled
  __shared__ float sWk2[64];
  __shared__ __align__(16) unsigned int sT[8][512]; // per-wave 16x64 bf16 tile
  const int tid = threadIdx.x;

  for (int d = tid; d < 4096; d += 256) {
    const int j  = d & 7;
    const int ln = (d >> 3) & 31;
    const int r  = d >> 8;
    const int ks = r & 3;
    const int nt = r >> 2;
    const int n  = nt * 16 + (ln & 15);
    const int k  = ks * 32 + ((ln >> 4) << 4) + 2 * j;
    sC1[d] = pack2bf(Wc1[k * 64 + n], Wc1[(k + 1) * 64 + n]);
    sK1[d] = pack2bf(Wk1[k * 64 + n], Wk1[(k + 1) * 64 + n]);
  }
  for (int d = tid; d < 1024; d += 256) {
    const int j  = d & 7;
    const int ln = (d >> 3) & 31;
    const int r  = d >> 8;
    const int ks = r & 1;
    const int nt = r >> 1;
    const int n  = nt * 16 + (ln & 15);
    const int k  = ks * 32 + ((ln >> 4) << 4) + 2 * j;
    sC2[d] = pack2bf(Wc2[k * 32 + n], Wc2[(k + 1) * 32 + n]);
  }
  if (tid < 64) sWk2[tid] = Wk2[tid];
  __syncthreads();

  const int wave = tid >> 5;
  const int lane = tid & 31;
  const int m    = lane & 15;
  const int half = lane >> 4;
  const int row0 = blockIdx.x * 128 + wave * 16;
  int row = row0 + m;
  if (row > Nrows - 1) row = Nrows - 1;
  const float2* hrow = reinterpret_cast<const float2*>(H + (long long)row * 128);

  FragBF a[4];
#pragma unroll
  for (int ks = 0; ks < 4; ++ks) {
#pragma unroll
    for (int j = 0; j < 8; ++j) {
      const int k = ks * 32 + half * 16 + 2 * j;
      float2 p = hrow[k >> 1];
      a[ks].u[j] = pack2bf(p.x, p.y);
    }
  }

  const bool full = (row0 + 16) <= Nrows;           // wave-uniform fast path
  unsigned int* T = sT[wave];
#pragma unroll
  for (int br = 0; br < 2; ++br) {
    const unsigned int* sW1 = br ? sK1 : sC1;
    const float* b1 = br ? bk1 : bc1;
    // stage 1: t = relu(H @ W1 + b1)  -> per-wave LDS as bf16
#pragma unroll
    for (int nt = 0; nt < 4; ++nt) {
      v8f acc = {0.f, 0.f, 0.f, 0.f, 0.f, 0.f, 0.f, 0.f};
#pragma unroll
      for (int ks = 0; ks < 4; ++ks) {
        FragBF b;
        const uint4* p = reinterpret_cast<const uint4*>(&sW1[((nt * 4 + ks) * 32 + lane) * 8]);
        b.q[0] = p[0];
        b.q[1] = p[1];
        acc = __builtin_amdgcn_wmma_f32_16x16x32_bf16(false, a[ks].v, false, b.v,
                                                      (short)0, acc, false, false);
      }
      const int n = nt * 16 + m;
      const float bv = b1[n];
      unsigned short* Ts = reinterpret_cast<unsigned short*>(T);
#pragma unroll
      for (int vi = 0; vi < 8; ++vi) {
        float t = acc[vi] + bv;
        t = (t > 0.f) ? t : 0.f;
        Ts[(vi + half * 8) * 64 + n] = f2bf(t);
      }
    }
    if (br == 0) {
      // stage 2: class = t @ Wc2 + bc2   (K=64 -> 2 WMMA k-steps, 2 n-tiles)
      FragBF a2[2];
#pragma unroll
      for (int ks = 0; ks < 2; ++ks) {
#pragma unroll
        for (int j = 0; j < 8; ++j) {
          const int k = ks * 32 + half * 16 + 2 * j;
          a2[ks].u[j] = T[m * 32 + (k >> 1)];
        }
      }
#pragma unroll
      for (int nt = 0; nt < 2; ++nt) {
        v8f acc = {0.f, 0.f, 0.f, 0.f, 0.f, 0.f, 0.f, 0.f};
#pragma unroll
        for (int ks = 0; ks < 2; ++ks) {
          FragBF b;
          const uint4* p = reinterpret_cast<const uint4*>(&sC2[((nt * 2 + ks) * 32 + lane) * 8]);
          b.q[0] = p[0];
          b.q[1] = p[1];
          acc = __builtin_amdgcn_wmma_f32_16x16x32_bf16(false, a2[ks].v, false, b.v,
                                                        (short)0, acc, false, false);
        }
        const int n = nt * 16 + m;
        const float bv = bc2[n];
        float* ob = cls + (long long)(row0 + half * 8) * 32 + n;
        if (full) {
#pragma unroll
          for (int vi = 0; vi < 8; ++vi) ob[vi * 32] = acc[vi] + bv;
        } else {
#pragma unroll
          for (int vi = 0; vi < 8; ++vi) {
            if (row0 + vi + half * 8 < Nrows) ob[vi * 32] = acc[vi] + bv;
          }
        }
      }
    } else {
      // key head: 64 -> 1, lanes 0..15 each own one row
      if (half == 0) {
        const unsigned short* Ts = reinterpret_cast<const unsigned short*>(T);
        float s = bk2[0];
#pragma unroll 8
        for (int k = 0; k < 64; ++k) s += bf2f(Ts[m * 64 + k]) * sWk2[k];
        const int rr = row0 + m;
        if (rr < Nrows) key[rr] = s;
      }
    }
  }
}

// ---------------------------------------------------------------------------
extern "C" void kernel_launch(void* const* d_in, const int* in_sizes, int n_in,
                              void* d_out, int out_size, void* d_ws, size_t ws_size,
                              hipStream_t stream) {
  (void)n_in; (void)out_size; (void)ws_size;
  const float*     x   = (const float*)d_in[0];
  const long long* ei  = (const long long*)d_in[1];   // int64 [2,E]
  const float*     ew  = (const float*)d_in[2];
  const float*     W0  = (const float*)d_in[3];
  const float*     b0  = (const float*)d_in[4];
  const float*     W1  = (const float*)d_in[5];
  const float*     b1  = (const float*)d_in[6];
  const float *gamma0 = (const float*)d_in[7],  *beta0 = (const float*)d_in[8];
  const float *mean0  = (const float*)d_in[9],  *var0  = (const float*)d_in[10];
  const float *gamma1 = (const float*)d_in[11], *beta1 = (const float*)d_in[12];
  const float *mean1  = (const float*)d_in[13], *var1  = (const float*)d_in[14];
  const float *Wc1 = (const float*)d_in[15], *bc1 = (const float*)d_in[16];
  const float *Wc2 = (const float*)d_in[17], *bc2 = (const float*)d_in[18];
  const float *Wk1 = (const float*)d_in[19], *bk1 = (const float*)d_in[20];
  const float *Wk2 = (const float*)d_in[21], *bk2 = (const float*)d_in[22];

  const int N = in_sizes[0] / 128;
  const int E = in_sizes[2];

  float* out = (float*)d_out;
  float* cls = out;                                // [N,32]
  float* key = out + (size_t)N * 32;               // [N,1]
  float* emb = out + (size_t)N * 33;               // [N,128]

  float* ws = (float*)d_ws;
  size_t o = 0;
  float* dinv = ws + o;  o += (((size_t)N + 31) & ~(size_t)31);
  float* nrm  = ws + o;  o += (((size_t)E + 31) & ~(size_t)31);
  float* bufA = ws + o;  o += (size_t)N * 128;     // GEMM output / gather source
  float* bufB = ws + o;                            // accumulator / activations

  const int blkN    = (N + 255) / 256;
  const int blkE    = (E + 255) / 256;
  const int blkRow  = (N + 127) / 128;
  const int nvec    = N * 32;                      // N*128 floats as float4
  const int blkV    = (nvec + 255) / 256;
  const int blkEdge = (E + 7) / 8;                 // one wave32 per edge

  // Graph normalization (shared by both layers)
  gnn_deg_init<<<blkN, 256, 0, stream>>>(dinv, N);
  gnn_deg_acc <<<blkE, 256, 0, stream>>>(ei, ew, dinv, E);
  gnn_dinv    <<<blkN, 256, 0, stream>>>(dinv, N);
  gnn_norm    <<<blkE, 256, 0, stream>>>(ei, ew, dinv, nrm, E);

  // Layer 0: xW0 -> aggregate -> bias+BN+ReLU (in place)
  gnn_gemm128 <<<blkRow,  256, 0, stream>>>(x, W0, bufA, N);
  gnn_selfinit<<<blkV,    256, 0, stream>>>(bufA, dinv, bufB, nvec);
  gnn_edge_agg<<<blkEdge, 256, 0, stream>>>(bufA, nrm, ei, bufB, E);
  gnn_bnrelu  <<<blkV,    256, 0, stream>>>(bufB, b0, gamma0, beta0, mean0, var0, bufB, nvec);

  // Layer 1: h0@W1 -> aggregate -> bias+BN+ReLU -> node_embedding (d_out)
  gnn_gemm128 <<<blkRow,  256, 0, stream>>>(bufB, W1, bufA, N);
  gnn_selfinit<<<blkV,    256, 0, stream>>>(bufA, dinv, bufB, nvec);
  gnn_edge_agg<<<blkEdge, 256, 0, stream>>>(bufA, nrm, ei, bufB, E);
  gnn_bnrelu  <<<blkV,    256, 0, stream>>>(bufB, b1, gamma1, beta1, mean1, var1, emb, nvec);

  // Fused MLP heads
  gnn_heads<<<blkRow, 256, 0, stream>>>(emb, Wc1, bc1, Wc2, bc2,
                                        Wk1, bk1, Wk2, bk2, cls, key, N);
}
